// lastModule__40218073759903
// MI455X (gfx1250) — matrix operand order repaired and checked by reading, not compiled, not run on hardware
//
#include <hip/hip_runtime.h>

typedef __attribute__((ext_vector_type(16))) __bf16 v16bf;
typedef __attribute__((ext_vector_type(8)))  float  v8f;

#define B_SZ   1024
#define IN_F   512
#define O_F    64
#define E_N    8
#define K_MLP  512   // O_F * E_N

#define BM 128       // batch rows per workgroup
#define BN 128       // j-columns per outer step (8 WMMA n-tiles per wave)
#define BK 32        // i-slice per WMMA (bf16 K=32)
#define NT (BN / 16) // 8 n-tiles

#define XSTRIDE 17   // 16 uint pairs per 32-elem row + 1 pad (bank spread)
#define WSTRIDE 17

// round-to-nearest-even f32 -> bf16, packed pair
__device__ __forceinline__ unsigned int pack_bf16x2(float lo, float hi) {
  unsigned int ul = __float_as_uint(lo);
  unsigned int uh = __float_as_uint(hi);
  ul += 0x7FFFu + ((ul >> 16) & 1u);
  uh += 0x7FFFu + ((uh >> 16) & 1u);
  return (ul >> 16) | (uh & 0xFFFF0000u);
}

__global__ __launch_bounds__(256)
void ensemble_mlp_wmma_kernel(const float* __restrict__ X,
                              const float* __restrict__ W1,
                              const float* __restrict__ b1,
                              const float* __restrict__ W2,
                              const float* __restrict__ b2,
                              float* __restrict__ out) {
  __shared__ unsigned int lds_x[BM * XSTRIDE];  // X tile  [128 x 32] bf16 pairs (~8.5 KB)
  __shared__ unsigned int lds_w[BN * WSTRIDE];  // W1 tile [128 x 32] bf16 pairs (~8.5 KB)

  const int tid  = threadIdx.x;
  const int lane = tid & 31;
  const int ln   = lane & 15;     // column / row-in-strip
  const int hi   = lane >> 4;     // lane half selects K-group / M-group
  const int ms   = tid >> 5;      // wave id = m-strip (0..7)

  const int k     = blockIdx.x >> 3;  // MLP index (slow-varying -> L2 reuse of W1[k])
  const int mb    = blockIdx.x & 7;   // batch block
  const int mbase = mb * BM;

  // A-matrix 16x32 bf16 pair indices (ISA 7.12.2): V0..3 -> K{0..7}+8*hi, V4..7 -> K{16..23}+8*hi
  // B-matrix 32x16 bf16 pair indices: same K order as A: V0..3 -> K{0..7}+8*hi, V4..7 -> K{16..23}+8*hi
  int kpA[8], kpB[8];
#pragma unroll
  for (int v = 0; v < 8; ++v) {
    kpA[v] = ((v < 4) ? v : v + 4) + hi * 4;
    kpB[v] = ((v < 4) ? v : v + 4) + hi * 4;
  }

  const long wbase = (long)k * IN_F * IN_F;  // W1[k]
  const long kbase = (long)k * IN_F;         // b1[k], W2[k]

  float acc[8];
#pragma unroll
  for (int r = 0; r < 8; ++r) acc[r] = 0.0f;

  for (int jt = 0; jt < IN_F; jt += BN) {   // 4 outer steps
    v8f c[NT];
#pragma unroll
    for (int t = 0; t < NT; ++t)
#pragma unroll
      for (int r = 0; r < 8; ++r) c[t][r] = 0.0f;

    for (int it = 0; it < IN_F; it += BK) {  // 16 k-steps
      __syncthreads();  // previous tile fully consumed

      // stage X tile [BM x BK] f32 -> bf16 (coalesced b128 loads), 4 float4/thread
#pragma unroll
      for (int q = 0; q < 4; ++q) {
        int fid  = q * 256 + tid;   // 1024 float4s
        int row  = fid >> 3;        // 8 float4 per 32-elem row
        int col4 = fid & 7;
        float4 xv = *(const float4*)(X + (long)(mbase + row) * IN_F + it + col4 * 4);
        lds_x[row * XSTRIDE + col4 * 2 + 0] = pack_bf16x2(xv.x, xv.y);
        lds_x[row * XSTRIDE + col4 * 2 + 1] = pack_bf16x2(xv.z, xv.w);
      }
      // stage W1 tile [BN x BK] f32 -> bf16; B(K=i, N=j) = W1[k][j][i], i contiguous
#pragma unroll
      for (int q = 0; q < 4; ++q) {
        int fid  = q * 256 + tid;   // 1024 float4s
        int row  = fid >> 3;        // j row (0..127)
        int col4 = fid & 7;
        float4 wv = *(const float4*)(W1 + wbase + (long)(jt + row) * IN_F + it + col4 * 4);
        lds_w[row * WSTRIDE + col4 * 2 + 0] = pack_bf16x2(wv.x, wv.y);
        lds_w[row * WSTRIDE + col4 * 2 + 1] = pack_bf16x2(wv.z, wv.w);
      }
      __syncthreads();

      // A fragment: amortized over 8 n-tile WMMAs
      union AU { unsigned int u[8]; v16bf v; } a;
      const int arow = (ms * 16 + ln) * XSTRIDE;
#pragma unroll
      for (int v = 0; v < 8; ++v) a.u[v] = lds_x[arow + kpA[v]];

#pragma unroll
      for (int t = 0; t < NT; ++t) {
        union BU { unsigned int u[8]; v16bf v; } bmx;
        const int brow = (t * 16 + ln) * WSTRIDE;
#pragma unroll
        for (int v = 0; v < 8; ++v) bmx.u[v] = lds_w[brow + kpB[v]];
        c[t] = __builtin_amdgcn_wmma_f32_16x16x32_bf16(
                   false, a.v, false, bmx.v, (short)0, c[t], false, false);
      }
    }

    // fused epilogue: h = relu(c + b1[k,j]); acc += h * W2[k,j]  (j = this lane's column)
#pragma unroll
    for (int t = 0; t < NT; ++t) {
      const int j = jt + t * 16 + ln;
      const float bv = b1[kbase + j];
      const float wv = W2[kbase + j];
#pragma unroll
      for (int r = 0; r < 8; ++r) {
        float h = c[t][r] + bv;
        h = fmaxf(h, 0.0f);
        acc[r] += h * wv;
      }
    }
  }

  // finish j-reduction: butterfly over the 16 lanes of each half-wave
#pragma unroll
  for (int r = 0; r < 8; ++r) {
#pragma unroll
    for (int m = 8; m >= 1; m >>= 1)
      acc[r] += __shfl_xor(acc[r], m, 16);
  }

  if (ln == 0) {
    const int o = k >> 3, e = k & 7;   // k = o*E + e
    const float bias2 = b2[k];
#pragma unroll
    for (int r = 0; r < 8; ++r) {
      const int brow = mbase + ms * 16 + hi * 8 + r;  // C-layout row M = r + 8*hi
      out[(long)brow * (O_F * E_N) + e * O_F + o] = acc[r] + bias2;
    }
  }
}

extern "C" void kernel_launch(void* const* d_in, const int* in_sizes, int n_in,
                              void* d_out, int out_size, void* d_ws, size_t ws_size,
                              hipStream_t stream) {
  (void)in_sizes; (void)n_in; (void)out_size; (void)d_ws; (void)ws_size;
  const float* X  = (const float*)d_in[0];
  const float* W1 = (const float*)d_in[1];
  const float* b1 = (const float*)d_in[2];
  const float* W2 = (const float*)d_in[3];
  const float* b2 = (const float*)d_in[4];
  float* out = (float*)d_out;

  dim3 grid(K_MLP * (B_SZ / BM));  // 512 k-slices x 8 batch blocks = 4096 WGs
  dim3 block(256);                 // 8 wave32 waves
  hipLaunchKernelGGL(ensemble_mlp_wmma_kernel, grid, block, 0, stream,
                     X, W1, b1, W2, b2, out);
}
